// AttnAdaINCos_163208757940
// MI455X (gfx1250) — matrix operand order repaired and checked by reading, not compiled, not run on hardware
//
#include <hip/hip_runtime.h>
#include <hip/hip_bf16.h>
#include <math.h>

typedef __bf16 bf16_t;
typedef __attribute__((ext_vector_type(2)))  __bf16 v2bf;
typedef __attribute__((ext_vector_type(4)))  __bf16 v4bf;
typedef __attribute__((ext_vector_type(8)))  __bf16 v8bf;
typedef __attribute__((ext_vector_type(16))) __bf16 v16bf;
typedef __attribute__((ext_vector_type(4)))  float  v4f;
typedef __attribute__((ext_vector_type(8)))  float  v8f;

#define EPSN 1e-5f
#define CHN 512
#define PIX 4096
#define NBATCH 4

static __device__ __forceinline__ v16bf cat8(v8bf lo, v8bf hi) {
  return __builtin_shufflevector(lo, hi, 0,1,2,3,4,5,6,7,8,9,10,11,12,13,14,15);
}

// A fragment (16x32 bf16): src row-major [16+][ld], logical A[m][k].
// Lane: m = lane%16, h = lane/16. Element i: k = (i/8)*16 + h*8 + (i%8).
static __device__ __forceinline__ v16bf frag_a(const bf16_t* src, int ld) {
  int lane = threadIdx.x & 31;
  int m = lane & 15, h = lane >> 4;
  const bf16_t* r = src + m * ld + h * 8;
  return cat8(*(const v8bf*)r, *(const v8bf*)(r + 16));
}

// B fragment (32x16 bf16): src row-major [16 n][ld] holding B^T (src[n][k] = B[k][n]).
// Lane: n = lane%16, h = lane/16. Element i: k = (i/8)*16 + h*8 + (i%8).
static __device__ __forceinline__ v16bf frag_b(const bf16_t* src, int ld) {
  int lane = threadIdx.x & 31;
  int n = lane & 15, h = lane >> 4;
  const bf16_t* r = src + n * ld + h * 8;
  return cat8(*(const v8bf*)r, *(const v8bf*)(r + 16));
}

static __device__ __forceinline__ v8f wmma_bf16(v16bf a, v16bf b, v8f c) {
  return __builtin_amdgcn_wmma_f32_16x16x32_bf16(false, a, false, b, (short)0, c, false, false);
}

static __device__ __forceinline__ v8f vzero8() {
  v8f z = {0.f, 0.f, 0.f, 0.f, 0.f, 0.f, 0.f, 0.f};
  return z;
}

// CDNA5 async global->LDS 16B copy (ASYNCcnt-tracked, per CDNA5 ISA).
// lds generic pointer: low 32 bits are the LDS byte offset (aperture addressing).
static __device__ __forceinline__ void async_copy_b128(void* lds, const void* gsrc) {
  unsigned loff = (unsigned)(unsigned long long)lds;
  unsigned long long ga = (unsigned long long)gsrc;
  asm volatile("global_load_async_to_lds_b128 %0, %1, off"
               :: "v"(loff), "v"(ga) : "memory");
}
static __device__ __forceinline__ void async_wait0() {
  asm volatile("s_wait_asynccnt 0" ::: "memory");
}

// ---------------------------------------------------------------------------
// 1x1 conv as GEMM: out[p][o] = sum_c X[b][c][p] * W[o][c] + bias[o]
// MODE 0: Y[b][p][o] bf16 ; MODE 1: Y[b][o][p], Y2[b][o][p] = Y^2 bf16
// ---------------------------------------------------------------------------
template <int MODE>
__global__ __launch_bounds__(256)
void conv1x1_kernel(const float* __restrict__ X, const float* __restrict__ W,
                    const float* __restrict__ bias,
                    bf16_t* __restrict__ Y, bf16_t* __restrict__ Y2)
{
  __shared__ __align__(16) bf16_t Al[64 * 40]; // [pixel][k]
  __shared__ __align__(16) bf16_t Bl[64 * 40]; // [o][k]  (B^T layout)
  __shared__ __align__(16) bf16_t Dl[64 * 72]; // staging for coalesced output
  const int t = threadIdx.x;
  const int w = t >> 5;
  const int lane = t & 31;
  const int n = lane & 15, h = lane >> 4;
  const int p0 = blockIdx.x * 64;
  const int o0 = blockIdx.y * 64;
  const int bi = blockIdx.z;
  const size_t xbase = (size_t)bi * CHN * PIX;

  const int tr  = w & 3;          // 16-row (pixel) block
  const int tcb = (w >> 2) * 2;   // two 16-col (out-ch) blocks

  v8f acc[2] = {vzero8(), vzero8()};

  for (int c0 = 0; c0 < CHN; c0 += 32) {
    #pragma unroll
    for (int j = 0; j < 2; ++j) {               // A tile 64p x 32k, 4-wide packs
      int q = t + 256 * j;                      // 512 quads
      int p = q & 63, k = (q >> 6) * 4;
      v4bf pk;
      #pragma unroll
      for (int e = 0; e < 4; ++e)
        pk[e] = (bf16_t)X[xbase + (size_t)(c0 + k + e) * PIX + p0 + p];
      *(v4bf*)&Al[p * 40 + k] = pk;
    }
    #pragma unroll
    for (int j = 0; j < 2; ++j) {               // B tile 64o x 32k, float4 reads
      int q = t + 256 * j;
      int k = (q & 7) * 4, o = q >> 3;
      v4f wv = *(const v4f*)&W[(size_t)(o0 + o) * CHN + c0 + k];
      v4bf pk;
      #pragma unroll
      for (int e = 0; e < 4; ++e) pk[e] = (bf16_t)wv[e];
      *(v4bf*)&Bl[o * 40 + k] = pk;
    }
    __syncthreads();
    v16bf a  = frag_a(Al + tr * 16 * 40, 40);
    v16bf b0 = frag_b(Bl + (tcb + 0) * 16 * 40, 40);
    v16bf b1 = frag_b(Bl + (tcb + 1) * 16 * 40, 40);
    acc[0] = wmma_bf16(a, b0, acc[0]);
    acc[1] = wmma_bf16(a, b1, acc[1]);
    __syncthreads();
  }

  float bv[2] = {bias[o0 + tcb * 16 + n], bias[o0 + (tcb + 1) * 16 + n]};

  if (MODE == 0) {
    #pragma unroll
    for (int tc = 0; tc < 2; ++tc) {
      int ocol = (tcb + tc) * 16 + n;
      #pragma unroll
      for (int r = 0; r < 8; ++r)
        Dl[(tr * 16 + h * 8 + r) * 72 + ocol] = (bf16_t)(acc[tc][r] + bv[tc]);
    }
    __syncthreads();
    #pragma unroll
    for (int j = 0; j < 2; ++j) {
      int c2 = t + 256 * j;
      int pl = c2 >> 3, ch = c2 & 7;
      *(v8bf*)&Y[((size_t)bi * PIX + p0 + pl) * CHN + o0 + ch * 8] =
          *(const v8bf*)&Dl[pl * 72 + ch * 8];
    }
  } else {
    #pragma unroll
    for (int tc = 0; tc < 2; ++tc) {
      int ocol = (tcb + tc) * 16 + n;
      #pragma unroll
      for (int r = 0; r < 8; ++r)
        Dl[ocol * 72 + tr * 16 + h * 8 + r] = (bf16_t)(acc[tc][r] + bv[tc]);
    }
    __syncthreads();
    #pragma unroll
    for (int j = 0; j < 2; ++j) {
      int c2 = t + 256 * j;
      int ol = c2 >> 3, ch = c2 & 7;
      *(v8bf*)&Y[((size_t)bi * CHN + o0 + ol) * PIX + p0 + ch * 8] =
          *(const v8bf*)&Dl[ol * 72 + ch * 8];
    }
    __syncthreads();
    #pragma unroll
    for (int tc = 0; tc < 2; ++tc) {
      int ocol = (tcb + tc) * 16 + n;
      #pragma unroll
      for (int r = 0; r < 8; ++r) {
        float v = acc[tc][r] + bv[tc];
        Dl[ocol * 72 + tr * 16 + h * 8 + r] = (bf16_t)(v * v);
      }
    }
    __syncthreads();
    #pragma unroll
    for (int j = 0; j < 2; ++j) {
      int c2 = t + 256 * j;
      int ol = c2 >> 3, ch = c2 & 7;
      *(v8bf*)&Y2[((size_t)bi * CHN + o0 + ol) * PIX + p0 + ch * 8] =
          *(const v8bf*)&Dl[ol * 72 + ch * 8];
    }
  }
}

// ---------------------------------------------------------------------------
// rnorm[row] = 1 / (||Y[row][:]||_2 + eps)
// ---------------------------------------------------------------------------
__global__ __launch_bounds__(256)
void rownorm_kernel(const bf16_t* __restrict__ Y, float* __restrict__ rnorm)
{
  int gw = (blockIdx.x * 256 + threadIdx.x) >> 5;
  int lane = threadIdx.x & 31;
  const bf16_t* row = Y + (size_t)gw * CHN + lane * 16;
  float s = 0.f;
  #pragma unroll
  for (int j = 0; j < 2; ++j) {
    v8bf v = *(const v8bf*)(row + j * 8);
    #pragma unroll
    for (int e = 0; e < 8; ++e) { float f = (float)v[e]; s += f * f; }
  }
  #pragma unroll
  for (int off = 16; off > 0; off >>= 1) s += __shfl_xor(s, off, 32);
  if (lane == 0) rnorm[gw] = 1.0f / (sqrtf(s) + EPSN);
}

// ---------------------------------------------------------------------------
// Fused attention: 64-row tiles, 16 waves. Phase 1: S = relu(QK^T*qs*ks + 1)
// (one 16x16 WMMA fragment per wave), rowsums; Phase 2: each wave owns a
// unique 64-wide slice of [V | V^2] -> acc += S @ V  (no V re-read redundancy).
// ---------------------------------------------------------------------------
#define LDQ 520
#define LDSS 72

__global__ __launch_bounds__(512)
void attn_kernel(const bf16_t* __restrict__ Fb,  // [B][4096][512]
                 const bf16_t* __restrict__ Gb,  // [B][4096][512]
                 const bf16_t* __restrict__ Vt,  // [B][512][4096]
                 const bf16_t* __restrict__ V2t, // [B][512][4096]
                 const float* __restrict__ rF, const float* __restrict__ rG,
                 float* __restrict__ meanU,      // [B][4096][512]
                 float* __restrict__ m2U,        // [B][4096][512]
                 float* __restrict__ rowsum)     // [B*4096]
{
  __shared__ __align__(16) bf16_t Ql[64 * LDQ];   // Q tile; reused for epilogue
  __shared__ __align__(16) bf16_t Sl[64 * LDSS];  // S tile (bf16)
  __shared__ float rs[64];
  __shared__ float qs[64];

  const int t = threadIdx.x;
  const int w = t >> 5;                           // 16 waves
  const int lane = t & 31;
  const int n = lane & 15, h = lane >> 4;
  const int p0 = blockIdx.x * 64;
  const int bi = blockIdx.y;

  // stage Q tile 64x512 via CDNA5 async global->LDS b128 copies
  #pragma unroll
  for (int j = 0; j < 8; ++j) {
    int c = t + 512 * j;                          // 4096 16B chunks
    int row = c >> 6, koff = (c & 63) * 8;
    async_copy_b128(&Ql[row * LDQ + koff],
                    &Fb[((size_t)bi * PIX + p0 + row) * CHN + koff]);
  }
  if (t < 64) { qs[t] = rF[bi * PIX + p0 + t]; rs[t] = 0.f; }
  async_wait0();
  __syncthreads();

  // phase-2 ownership: 16 waves x 64 channels over [V(512) | V^2(512)]
  const int obase = (w & 7) * 64;
  const bf16_t* Vsrc = (w < 8) ? Vt : V2t;
  v8f acc[4][4];
  #pragma unroll
  for (int i = 0; i < 4; ++i)
    #pragma unroll
    for (int j = 0; j < 4; ++j) acc[i][j] = vzero8();

  // phase-1 fragment: S tile 64x64 = 4x4 fragments, one per wave
  const int srb = w >> 2;                         // row block 0..3
  const int scb = w & 3;                          // col block 0..3

  for (int m0 = 0; m0 < PIX; m0 += 64) {
    // ---- phase 1: S fragment ----
    v8f sacc = vzero8();
    const bf16_t* gbase = Gb + ((size_t)bi * PIX + m0 + scb * 16) * CHN;
    // prefetch next tile's K rows (global_prefetch_b8, speculative)
    __builtin_prefetch(gbase + (size_t)64 * CHN, 0, 1);
    #pragma unroll
    for (int c0 = 0; c0 < CHN; c0 += 32) {
      v16bf a = frag_a(Ql + srb * 16 * LDQ + c0, LDQ);
      v16bf b = frag_b(gbase + c0, CHN);
      sacc = wmma_bf16(a, b, sacc);
    }
    float ks = rG[bi * PIX + m0 + scb * 16 + n];
    float rp[8];
    #pragma unroll
    for (int r = 0; r < 8; ++r) {
      int row = srb * 16 + h * 8 + r;
      float s = fmaxf(sacc[r] * qs[row] * ks + 1.0f, 0.0f);
      rp[r] = s;
      Sl[row * LDSS + scb * 16 + n] = (bf16_t)s;
    }
    #pragma unroll
    for (int r = 0; r < 8; ++r) {
      #pragma unroll
      for (int off = 1; off < 16; off <<= 1) rp[r] += __shfl_xor(rp[r], off, 32);
    }
    if (n == 0) {
      #pragma unroll
      for (int r = 0; r < 8; ++r) atomicAdd(&rs[srb * 16 + h * 8 + r], rp[r]);
    }
    __syncthreads();

    // ---- phase 2: acc += S_tile(64x64) @ V_slice(64x64) ----
    #pragma unroll
    for (int kk = 0; kk < 2; ++kk) {
      v16bf a[4];
      #pragma unroll
      for (int rb = 0; rb < 4; ++rb)
        a[rb] = frag_a(Sl + rb * 16 * LDSS + kk * 32, LDSS);
      #pragma unroll
      for (int cb = 0; cb < 4; ++cb) {
        const bf16_t* vb =
            Vsrc + ((size_t)bi * CHN + obase + cb * 16) * PIX + m0 + kk * 32;
        v16bf b = frag_b(vb, PIX);
        #pragma unroll
        for (int rb = 0; rb < 4; ++rb) acc[rb][cb] = wmma_bf16(a[rb], b, acc[rb][cb]);
      }
    }
    __syncthreads();
  }

  // epilogue: stage per-wave f32 tiles through retired Ql, write coalesced b128
  float* dst = (w < 8) ? meanU : m2U;
  float* buf = (float*)Ql + w * 1024;             // 16 waves x 4KB <= 64KB
  #pragma unroll
  for (int rb = 0; rb < 4; ++rb) {
    #pragma unroll
    for (int cb = 0; cb < 4; ++cb)
      #pragma unroll
      for (int r = 0; r < 8; ++r)
        buf[(h * 8 + r) * 64 + cb * 16 + n] = acc[rb][cb][r];
    __syncthreads();
    #pragma unroll
    for (int jj = 0; jj < 8; ++jj) {
      int c = lane + 32 * jj;                     // 256 16B chunks (16 rows x 64 f32)
      int row = c >> 4, ch = c & 15;
      *(v4f*)&dst[((size_t)bi * PIX + p0 + rb * 16 + row) * CHN + obase + ch * 4] =
          *(const v4f*)&buf[row * 64 + ch * 4];
    }
    __syncthreads();
  }
  if (t < 64) rowsum[bi * PIX + p0 + t] = rs[t];
}

// ---------------------------------------------------------------------------
// Per-(b,c) content mean + 1/sqrt(var_unbiased + eps)
// ---------------------------------------------------------------------------
__global__ __launch_bounds__(256)
void stats_kernel(const float* __restrict__ content,
                  float* __restrict__ cmean, float* __restrict__ cinv)
{
  __shared__ float s1[256], s2[256];
  int ch = blockIdx.x;
  const float* src = content + (size_t)ch * PIX;
  int t = threadIdx.x;
  float sum = 0.f, sq = 0.f;
  #pragma unroll
  for (int i = 0; i < 4; ++i) {
    v4f v = *(const v4f*)&src[(t + 256 * i) * 4];
    #pragma unroll
    for (int e = 0; e < 4; ++e) { sum += v[e]; sq += v[e] * v[e]; }
  }
  s1[t] = sum; s2[t] = sq; __syncthreads();
  for (int off = 128; off > 0; off >>= 1) {
    if (t < off) { s1[t] += s1[t + off]; s2[t] += s2[t + off]; }
    __syncthreads();
  }
  if (t == 0) {
    float m = s1[0] / (float)PIX;
    float var = (s2[0] - (float)PIX * m * m) / (float)(PIX - 1);
    cmean[ch] = m;
    cinv[ch] = 1.0f / sqrtf(var + EPSN);
  }
}

// ---------------------------------------------------------------------------
// out[b][c][p] = std * (content - cmean)*cinv + mean   (LDS transpose tile)
// ---------------------------------------------------------------------------
__global__ __launch_bounds__(256)
void final_kernel(const float* __restrict__ content,
                  const float* __restrict__ meanU, const float* __restrict__ m2U,
                  const float* __restrict__ rowsum,
                  const float* __restrict__ cmean, const float* __restrict__ cinv,
                  float* __restrict__ out)
{
  __shared__ float meanT[64 * 65];
  __shared__ float stdT[64 * 65];
  __shared__ float rsi[64], cm[64], ci[64];
  int t = threadIdx.x;
  int c0 = blockIdx.x * 64, p0 = blockIdx.y * 64, bi = blockIdx.z;
  if (t < 64) rsi[t] = 1.0f / (rowsum[bi * PIX + p0 + t] + EPSN);
  if (t >= 64 && t < 128) {
    int c = t - 64;
    cm[c] = cmean[bi * CHN + c0 + c];
    ci[c] = cinv[bi * CHN + c0 + c];
  }
  __syncthreads();
  #pragma unroll
  for (int j = 0; j < 16; ++j) {                  // read [p][c] coalesced on c
    int idx = t + 256 * j;
    int pl = idx >> 6, cl = idx & 63;
    size_t off = ((size_t)bi * PIX + p0 + pl) * CHN + c0 + cl;
    float inv = rsi[pl];
    float mean = meanU[off] * inv;
    float msq  = m2U[off] * inv;
    meanT[cl * 65 + pl] = mean;
    stdT[cl * 65 + pl]  = sqrtf(fmaxf(msq - mean * mean, 0.0f));
  }
  __syncthreads();
  #pragma unroll
  for (int j = 0; j < 16; ++j) {                  // write [c][p] coalesced on p
    int idx = t + 256 * j;
    int cl = idx >> 6, pl = idx & 63;
    size_t off = ((size_t)bi * CHN + c0 + cl) * PIX + p0 + pl;
    float x = content[off];
    out[off] = stdT[cl * 65 + pl] * (x - cm[cl]) * ci[cl] + meanT[cl * 65 + pl];
  }
}

extern "C" void kernel_launch(void* const* d_in, const int* in_sizes, int n_in,
                              void* d_out, int out_size, void* d_ws, size_t ws_size,
                              hipStream_t stream)
{
  (void)in_sizes; (void)n_in; (void)out_size; (void)ws_size;
  const float* content = (const float*)d_in[0];
  const float* style   = (const float*)d_in[1];
  const float* ckey    = (const float*)d_in[2];
  const float* skey    = (const float*)d_in[3];
  const float* Wf = (const float*)d_in[4];
  const float* bf = (const float*)d_in[5];
  const float* Wg = (const float*)d_in[6];
  const float* bg = (const float*)d_in[7];
  const float* Wh = (const float*)d_in[8];
  const float* bh = (const float*)d_in[9];
  float* out = (float*)d_out;

  char* ws = (char*)d_ws;
  size_t off = 0;
  auto alloc = [&](size_t bytes) -> char* {
    char* p = ws + off;
    off = (off + bytes + 255) & ~(size_t)255;
    return p;
  };
  const size_t BP = (size_t)NBATCH * PIX;
  bf16_t* Fb  = (bf16_t*)alloc(BP * CHN * sizeof(bf16_t));
  bf16_t* Gb  = (bf16_t*)alloc(BP * CHN * sizeof(bf16_t));
  bf16_t* Vt  = (bf16_t*)alloc(BP * CHN * sizeof(bf16_t));
  bf16_t* V2t = (bf16_t*)alloc(BP * CHN * sizeof(bf16_t));
  float* rF     = (float*)alloc(BP * sizeof(float));
  float* rG     = (float*)alloc(BP * sizeof(float));
  float* meanU  = (float*)alloc(BP * CHN * sizeof(float));
  float* m2U    = (float*)alloc(BP * CHN * sizeof(float));
  float* rsum   = (float*)alloc(BP * sizeof(float));
  float* cmean  = (float*)alloc((size_t)NBATCH * CHN * sizeof(float));
  float* cinv   = (float*)alloc((size_t)NBATCH * CHN * sizeof(float));

  dim3 cgrid(PIX / 64, CHN / 64, NBATCH);
  conv1x1_kernel<0><<<cgrid, 256, 0, stream>>>(ckey, Wf, bf, Fb, nullptr);
  conv1x1_kernel<0><<<cgrid, 256, 0, stream>>>(skey, Wg, bg, Gb, nullptr);
  conv1x1_kernel<1><<<cgrid, 256, 0, stream>>>(style, Wh, bh, Vt, V2t);

  rownorm_kernel<<<BP / 8, 256, 0, stream>>>(Fb, rF);
  rownorm_kernel<<<BP / 8, 256, 0, stream>>>(Gb, rG);

  attn_kernel<<<dim3(PIX / 64, NBATCH), 512, 0, stream>>>(
      Fb, Gb, Vt, V2t, rF, rG, meanU, m2U, rsum);

  stats_kernel<<<NBATCH * CHN, 256, 0, stream>>>(content, cmean, cinv);

  final_kernel<<<dim3(CHN / 64, PIX / 64, NBATCH), 256, 0, stream>>>(
      content, meanU, m2U, rsum, cmean, cinv, out);
}
